// Model_824633721727
// MI455X (gfx1250) — matrix operand order repaired and checked by reading, not compiled, not run on hardware
//
#include <hip/hip_runtime.h>
#include <hip/hip_bf16.h>

// ---------------------------------------------------------------------------
// Problem constants (fixed by setup_inputs in the reference)
// ---------------------------------------------------------------------------
#define NN      10000
#define EE      160000
#define DD      512
#define LL      3
#define N_INV   4000
#define N_UNK   3000
#define NEG_SLOPE 0.2f
#define LN_EPS  1e-5f

typedef __bf16 bf16;
typedef __attribute__((ext_vector_type(16))) __bf16 v16bf;
typedef __attribute__((ext_vector_type(8)))  __bf16 v8bf;
typedef __attribute__((ext_vector_type(8)))  float  v8f;

// ---------------------------------------------------------------------------
// Small helpers
// ---------------------------------------------------------------------------
__device__ __forceinline__ float waveReduceSum(float v) {
#pragma unroll
  for (int off = 16; off > 0; off >>= 1) v += __shfl_xor(v, off, 32);
  return v;
}

__device__ __forceinline__ void atomicMaxFloat(float* addr, float v) {
  // classic sign-split trick: works for mixed-sign IEEE floats
  if (v >= 0.0f) atomicMax(reinterpret_cast<int*>(addr), __float_as_int(v));
  else           atomicMin(reinterpret_cast<unsigned int*>(addr), __float_as_uint(v));
}

// ---------------------------------------------------------------------------
// Elementwise conversion / init kernels
// ---------------------------------------------------------------------------
__global__ void f32_to_bf16_kernel(const float* __restrict__ src,
                                   bf16* __restrict__ dst, size_t n) {
  size_t i = (size_t)blockIdx.x * blockDim.x + threadIdx.x;
  size_t stride = (size_t)gridDim.x * blockDim.x;
  for (; i < n; i += stride) dst[i] = (bf16)src[i];
}

// src[R][C] (f32) -> dst[C][R] (bf16)
__global__ void f32_to_bf16_tr_kernel(const float* __restrict__ src,
                                      bf16* __restrict__ dst, int R, int C) {
  size_t n = (size_t)R * C;
  size_t i = (size_t)blockIdx.x * blockDim.x + threadIdx.x;
  size_t stride = (size_t)gridDim.x * blockDim.x;
  for (; i < n; i += stride) {
    int r = (int)(i / C);
    int c = (int)(i % C);
    dst[(size_t)c * R + r] = (bf16)src[i];
  }
}

__global__ void zero_f32_kernel(float* __restrict__ p, size_t n) {
  size_t i = (size_t)blockIdx.x * blockDim.x + threadIdx.x;
  size_t stride = (size_t)gridDim.x * blockDim.x;
  for (; i < n; i += stride) p[i] = 0.0f;
}

__global__ void init_softmax_kernel(float* __restrict__ smax,
                                    float* __restrict__ denom, int n) {
  int i = blockIdx.x * blockDim.x + threadIdx.x;
  if (i < n) { smax[i] = -3.0e38f; denom[i] = 0.0f; }
}

// concat [rows][512]|[rows][512] (f32) -> [rows][1024] bf16
__global__ void build_cat_kernel(const float* __restrict__ p0,
                                 const float* __restrict__ p1,
                                 bf16* __restrict__ dst, size_t rows) {
  size_t n = rows << 10;
  size_t i = (size_t)blockIdx.x * blockDim.x + threadIdx.x;
  size_t stride = (size_t)gridDim.x * blockDim.x;
  for (; i < n; i += stride) {
    size_t r = i >> 10;
    int c = (int)(i & 1023);
    float v = (c < DD) ? p0[(r << 9) + c] : p1[(r << 9) + (c - DD)];
    dst[i] = (bf16)v;
  }
}

// ---------------------------------------------------------------------------
// bf16 WMMA GEMM: C[M][N] = A[M][K] * B[K][N] (+ bias[N])
// A row-major bf16, B row-major (K-major) bf16, C f32.
// Block tile 128x128, BK=64, 8 waves (2x4), wave tile 64x32 (4x2 WMMA tiles).
// ---------------------------------------------------------------------------
#define LDSA 72    // 64 + 8 pad (bf16 elements)
#define LDSB 136   // 128 + 8 pad

__global__ __launch_bounds__(256)
void gemm_bf16_kernel(const bf16* __restrict__ A, int lda,
                      const bf16* __restrict__ B, int ldb,
                      float* __restrict__ C, int ldc,
                      const float* __restrict__ bias,
                      int M, int N, int K) {
  __shared__ bf16 As[128 * LDSA];
  __shared__ bf16 Bs[64 * LDSB];

  const int t    = threadIdx.x;
  const int lane = t & 31;
  const int wave = t >> 5;
  const int mw   = wave >> 2;    // 0..1
  const int nw   = wave & 3;     // 0..3
  const int m0   = blockIdx.y * 128;
  const int n0   = blockIdx.x * 128;

  v8f acc[4][2];
#pragma unroll
  for (int i = 0; i < 4; ++i)
#pragma unroll
    for (int j = 0; j < 2; ++j)
#pragma unroll
      for (int q = 0; q < 8; ++q) acc[i][j][q] = 0.0f;

  for (int k0 = 0; k0 < K; k0 += 64) {
    // ---- stage A tile (128 x 64) to LDS: 8 bf16 per thread per iter ----
#pragma unroll
    for (int it = 0; it < 4; ++it) {
      int idx = it * 256 + t;          // 0..1023 chunks of 8
      int row = idx >> 3;              // 0..127
      int col = (idx & 7) << 3;        // 0..56
      int gr  = m0 + row;
      v8bf v;
      if (gr < M) {
        v = *(const v8bf*)(A + (size_t)gr * lda + k0 + col);
        if (k0 + 64 < K)
          __builtin_prefetch(A + (size_t)gr * lda + k0 + 64 + col, 0, 0);
      } else {
#pragma unroll
        for (int q = 0; q < 8; ++q) v[q] = (bf16)0.0f;
      }
      *(v8bf*)(&As[row * LDSA + col]) = v;
    }
    // ---- stage B tile (64 x 128) to LDS (column-bounds guarded) ----
#pragma unroll
    for (int it = 0; it < 4; ++it) {
      int idx = it * 256 + t;
      int row = idx >> 4;              // 0..63
      int col = (idx & 15) << 3;       // 0..120
      const bf16* src = B + (size_t)(k0 + row) * ldb + n0 + col;
#pragma unroll
      for (int q = 0; q < 8; ++q) {
        bf16 v = (n0 + col + q < N) ? src[q] : (bf16)0.0f;
        Bs[row * LDSB + col + q] = v;
      }
    }
    __syncthreads();

    // ---- two K=32 WMMA steps over the staged tile ----
#pragma unroll
    for (int kk = 0; kk < 64; kk += 32) {
      v16bf af[4];
#pragma unroll
      for (int tm = 0; tm < 4; ++tm) {
        // A 16x32 bf16 fragment: lanes 0-15 -> M=lane, K {0..7,16..23};
        // lanes 16-31 -> M=lane-16, K {8..15,24..31}
        int r    = mw * 64 + tm * 16 + (lane & 15);
        int koff = kk + ((lane >> 4) << 3);
        const bf16* p = &As[r * LDSA + koff];
        v8bf lo = *(const v8bf*)(p);
        v8bf hi = *(const v8bf*)(p + 16);
#pragma unroll
        for (int q = 0; q < 8; ++q) { af[tm][q] = lo[q]; af[tm][q + 8] = hi[q]; }
      }
      v16bf bfg[2];
#pragma unroll
      for (int tn = 0; tn < 2; ++tn) {
        // B 32x16 bf16 fragment: lanes 0-15 -> N=lane, K {0..7,16..23};
        // lanes 16-31 -> N=lane-16, K {8..15,24..31}
        const bf16* p = &Bs[(kk + ((lane >> 4) << 3)) * LDSB + nw * 32 + tn * 16 + (lane & 15)];
#pragma unroll
        for (int q = 0; q < 8; ++q) { bfg[tn][q] = p[q * LDSB]; bfg[tn][q + 8] = p[(q + 16) * LDSB]; }
      }
#pragma unroll
      for (int tm = 0; tm < 4; ++tm)
#pragma unroll
        for (int tn = 0; tn < 2; ++tn)
          acc[tm][tn] = __builtin_amdgcn_wmma_f32_16x16x32_bf16(
              false, af[tm], false, bfg[tn], (short)0, acc[tm][tn], false, false);
    }
    __syncthreads();
  }

  // ---- store C (+bias); C/D layout: VGPR q -> M = base + q (+8 for hi lanes),
  //      N = base + (lane & 15) ----
#pragma unroll
  for (int tm = 0; tm < 4; ++tm) {
#pragma unroll
    for (int tn = 0; tn < 2; ++tn) {
      int mb = m0 + mw * 64 + tm * 16 + ((lane >> 4) << 3);
      int nc = n0 + nw * 32 + tn * 16 + (lane & 15);
      if (nc < N) {
        float bv = bias ? bias[nc] : 0.0f;
#pragma unroll
        for (int q = 0; q < 8; ++q) {
          int m = mb + q;
          if (m < M) C[(size_t)m * ldc + nc] = acc[tm][tn][q] + bv;
        }
      }
    }
  }
}

// ---------------------------------------------------------------------------
// GATv2 edge kernels (wave32 per edge, lane handles 16 of 512 features)
// ---------------------------------------------------------------------------
__global__ __launch_bounds__(256)
void edge_logits_kernel(const float* __restrict__ xl, const float* __restrict__ xr,
                        const float* __restrict__ ea, const float* __restrict__ We,
                        const float* __restrict__ att, const int* __restrict__ ei,
                        float* __restrict__ s) {
  int e = blockIdx.x * 8 + (threadIdx.x >> 5);
  int lane = threadIdx.x & 31;
  if (e >= EE) return;
  int src = ei[e], dst = ei[EE + e];
  float e0 = ea[2 * e], e1 = ea[2 * e + 1];
  const float* pl = xl + (size_t)src * DD;
  const float* pr = xr + (size_t)dst * DD;
  float acc = 0.0f;
#pragma unroll
  for (int j = 0; j < 16; ++j) {
    int c = lane + 32 * j;
    float m = pl[c] + pr[c] + e0 * We[c] + e1 * We[DD + c];
    m = (m > 0.0f) ? m : NEG_SLOPE * m;
    acc += att[c] * m;
  }
  acc = waveReduceSum(acc);
  if (lane == 0) s[e] = acc;
}

__global__ void edge_max_kernel(const float* __restrict__ s,
                                const int* __restrict__ ei,
                                float* __restrict__ smax) {
  int e = blockIdx.x * blockDim.x + threadIdx.x;
  if (e >= EE) return;
  atomicMaxFloat(&smax[ei[EE + e]], s[e]);
}

__global__ void edge_exp_kernel(const float* __restrict__ s,
                                const int* __restrict__ ei,
                                const float* __restrict__ smax,
                                float* __restrict__ aexp,
                                float* __restrict__ denom) {
  int e = blockIdx.x * blockDim.x + threadIdx.x;
  if (e >= EE) return;
  int dst = ei[EE + e];
  float a = __expf(s[e] - smax[dst]);
  aexp[e] = a;
  atomicAdd(&denom[dst], a);
}

__global__ __launch_bounds__(256)
void edge_scatter_kernel(const float* __restrict__ xl, const int* __restrict__ ei,
                         const float* __restrict__ aexp, const float* __restrict__ denom,
                         float* __restrict__ hc) {
  int e = blockIdx.x * 8 + (threadIdx.x >> 5);
  int lane = threadIdx.x & 31;
  if (e >= EE) return;
  int src = ei[e], dst = ei[EE + e];
  float den = denom[dst];
  float alpha = aexp[e] / ((den > 0.0f) ? den : 1.0f);
  const float* pl = xl + (size_t)src * DD;
  float* po = hc + (size_t)dst * DD;
#pragma unroll
  for (int j = 0; j < 16; ++j) {
    int c = lane + 32 * j;
    atomicAdd(&po[c], alpha * pl[c]);
  }
}

// h = relu(LN(hc + gat_bias + h)) ; one wave per row
__global__ __launch_bounds__(256)
void ln_relu_kernel(float* __restrict__ h, const float* __restrict__ hc,
                    const float* __restrict__ gbias, const float* __restrict__ g,
                    const float* __restrict__ b) {
  int row = blockIdx.x * 8 + (threadIdx.x >> 5);
  int lane = threadIdx.x & 31;
  if (row >= NN) return;
  const float* hr  = h  + (size_t)row * DD;
  const float* hcr = hc + (size_t)row * DD;
  float vals[16];
  float s = 0.0f;
#pragma unroll
  for (int j = 0; j < 16; ++j) {
    int c = lane + 32 * j;
    float v = hcr[c] + gbias[c] + hr[c];
    vals[j] = v;
    s += v;
  }
  float mean = waveReduceSum(s) * (1.0f / DD);
  float vs = 0.0f;
#pragma unroll
  for (int j = 0; j < 16; ++j) { float t = vals[j] - mean; vs += t * t; }
  float inv = rsqrtf(waveReduceSum(vs) * (1.0f / DD) + LN_EPS);
  float* ho = h + (size_t)row * DD;
#pragma unroll
  for (int j = 0; j < 16; ++j) {
    int c = lane + 32 * j;
    float o = (vals[j] - mean) * inv * g[c] + b[c];
    ho[c] = fmaxf(o, 0.0f);
  }
}

// ---------------------------------------------------------------------------
// Host-side orchestration
// ---------------------------------------------------------------------------
extern "C" void kernel_launch(void* const* d_in, const int* in_sizes, int n_in,
                              void* d_out, int out_size, void* d_ws, size_t ws_size,
                              hipStream_t stream) {
  (void)in_sizes; (void)n_in; (void)out_size; (void)ws_size;

  const float* x         = (const float*)d_in[0];
  const float* edge_attr = (const float*)d_in[1];
  const float* emb_inv_W = (const float*)d_in[2];
  const float* emb_inv_b = (const float*)d_in[3];
  const float* emb_org_W = (const float*)d_in[4];
  const float* emb_org_b = (const float*)d_in[5];
  const float* gat_Wl    = (const float*)d_in[6];
  const float* gat_bl    = (const float*)d_in[7];
  const float* gat_Wr    = (const float*)d_in[8];
  const float* gat_br    = (const float*)d_in[9];
  const float* gat_We    = (const float*)d_in[10];
  const float* gat_att   = (const float*)d_in[11];
  const float* gat_bias  = (const float*)d_in[12];
  const float* ln_g      = (const float*)d_in[13];
  const float* ln_b      = (const float*)d_in[14];
  const float* lin0_W    = (const float*)d_in[15];
  const float* lin0_b    = (const float*)d_in[16];
  const float* lin1_W    = (const float*)d_in[17];
  const float* lin1_b    = (const float*)d_in[18];
  const int*   edge_index= (const int*)d_in[19];
  float* out = (float*)d_out;

  // ---- workspace bump allocator (256B aligned) ----
  size_t off = 0;
  auto alloc = [&](size_t bytes) -> void* {
    void* p = (char*)d_ws + off;
    off += (bytes + 255) & ~(size_t)255;
    return p;
  };
  const size_t ND = (size_t)NN * DD;

  bf16* xb     = (bf16*)alloc(ND * 2);
  bf16* hb     = (bf16*)alloc(ND * 2);
  float* h0    = (float*)alloc(ND * 4);
  float* h     = (float*)alloc(ND * 4);
  float* xl    = (float*)alloc(ND * 4);
  float* xr    = (float*)alloc(ND * 4);
  float* hc    = (float*)alloc(ND * 4);
  float* s     = (float*)alloc((size_t)EE * 4);
  float* aexp  = (float*)alloc((size_t)EE * 4);
  float* smax  = (float*)alloc((size_t)NN * 4);
  float* denom = (float*)alloc((size_t)NN * 4);
  bf16* WembI  = (bf16*)alloc((size_t)DD * DD * 2);
  bf16* WembO  = (bf16*)alloc((size_t)DD * DD * 2);
  bf16* Wlb    = (bf16*)alloc((size_t)LL * DD * DD * 2);
  bf16* Wrb    = (bf16*)alloc((size_t)LL * DD * DD * 2);
  bf16* lin0b  = (bf16*)alloc((size_t)2 * DD * DD * 2);
  bf16* lin1b  = (bf16*)alloc((size_t)2 * DD * DD * 2);
  bf16* cat0b  = (bf16*)alloc((size_t)N_INV * 2 * DD * 2);
  bf16* cat1b  = (bf16*)alloc((size_t)N_UNK * 2 * DD * 2);
  float* y0    = (float*)alloc((size_t)N_INV * DD * 4);
  float* y1    = (float*)alloc((size_t)N_UNK * DD * 4);
  bf16* y0b    = (bf16*)alloc((size_t)N_INV * DD * 2);
  bf16* y1T    = (bf16*)alloc((size_t)DD * N_UNK * 2);

  const int CT = 256;          // elementwise threads
  const int CB = 2048;         // elementwise grid (grid-stride)
  auto gemm_grid = [](int M, int N) { return dim3((N + 127) / 128, (M + 127) / 128); };

  // ---- convert inputs / weights to bf16 ----
  f32_to_bf16_kernel<<<CB, CT, 0, stream>>>(x, xb, ND);
  f32_to_bf16_kernel<<<CB, CT, 0, stream>>>(emb_inv_W, WembI, (size_t)DD * DD);
  f32_to_bf16_kernel<<<CB, CT, 0, stream>>>(emb_org_W, WembO, (size_t)DD * DD);
  f32_to_bf16_kernel<<<CB, CT, 0, stream>>>(gat_Wl, Wlb, (size_t)LL * DD * DD);
  f32_to_bf16_kernel<<<CB, CT, 0, stream>>>(gat_Wr, Wrb, (size_t)LL * DD * DD);
  f32_to_bf16_kernel<<<CB, CT, 0, stream>>>(lin0_W, lin0b, (size_t)2 * DD * DD);
  f32_to_bf16_kernel<<<CB, CT, 0, stream>>>(lin1_W, lin1b, (size_t)2 * DD * DD);

  // ---- embeddings: h0[:4000] = x[:4000]@emb_inv_W+b ; rest with emb_org ----
  gemm_bf16_kernel<<<gemm_grid(N_INV, DD), 256, 0, stream>>>(
      xb, DD, WembI, DD, h0, DD, emb_inv_b, N_INV, DD, DD);
  gemm_bf16_kernel<<<gemm_grid(NN - N_INV, DD), 256, 0, stream>>>(
      xb + (size_t)N_INV * DD, DD, WembO, DD, h0 + (size_t)N_INV * DD, DD,
      emb_org_b, NN - N_INV, DD, DD);

  // h = h0
  hipMemcpyAsync(h, h0, ND * 4, hipMemcpyDeviceToDevice, stream);

  // ---- GATv2 layers ----
  for (int k = 0; k < LL; ++k) {
    f32_to_bf16_kernel<<<CB, CT, 0, stream>>>(h, hb, ND);
    gemm_bf16_kernel<<<gemm_grid(NN, DD), 256, 0, stream>>>(
        hb, DD, Wlb + (size_t)k * DD * DD, DD, xl, DD, gat_bl + k * DD, NN, DD, DD);
    gemm_bf16_kernel<<<gemm_grid(NN, DD), 256, 0, stream>>>(
        hb, DD, Wrb + (size_t)k * DD * DD, DD, xr, DD, gat_br + k * DD, NN, DD, DD);

    init_softmax_kernel<<<(NN + CT - 1) / CT, CT, 0, stream>>>(smax, denom, NN);
    zero_f32_kernel<<<CB, CT, 0, stream>>>(hc, ND);

    edge_logits_kernel<<<(EE + 7) / 8, 256, 0, stream>>>(
        xl, xr, edge_attr, gat_We + (size_t)k * 2 * DD, gat_att + k * DD,
        edge_index, s);
    edge_max_kernel<<<(EE + CT - 1) / CT, CT, 0, stream>>>(s, edge_index, smax);
    edge_exp_kernel<<<(EE + CT - 1) / CT, CT, 0, stream>>>(s, edge_index, smax, aexp, denom);
    edge_scatter_kernel<<<(EE + 7) / 8, 256, 0, stream>>>(xl, edge_index, aexp, denom, hc);

    ln_relu_kernel<<<(NN + 7) / 8, 256, 0, stream>>>(
        h, hc, gat_bias + k * DD, ln_g + k * DD, ln_b + k * DD);
  }

  // ---- final projections ----
  build_cat_kernel<<<CB, CT, 0, stream>>>(h0, h, cat0b, (size_t)N_INV);
  build_cat_kernel<<<CB, CT, 0, stream>>>(h0 + (size_t)(NN - N_UNK) * DD,
                                          h + (size_t)(NN - N_UNK) * DD,
                                          cat1b, (size_t)N_UNK);
  gemm_bf16_kernel<<<gemm_grid(N_INV, DD), 256, 0, stream>>>(
      cat0b, 2 * DD, lin0b, DD, y0, DD, lin0_b, N_INV, DD, 2 * DD);
  gemm_bf16_kernel<<<gemm_grid(N_UNK, DD), 256, 0, stream>>>(
      cat1b, 2 * DD, lin1b, DD, y1, DD, lin1_b, N_UNK, DD, 2 * DD);

  // ---- out = y0 @ y1^T ----
  f32_to_bf16_kernel<<<CB, CT, 0, stream>>>(y0, y0b, (size_t)N_INV * DD);
  f32_to_bf16_tr_kernel<<<CB, CT, 0, stream>>>(y1, y1T, N_UNK, DD);  // -> [DD][N_UNK]
  gemm_bf16_kernel<<<gemm_grid(N_INV, N_UNK), 256, 0, stream>>>(
      y0b, DD, y1T, N_UNK, out, N_UNK, nullptr, N_INV, N_UNK, DD);
}
